// Int4TinyLSTM_86560770884282
// MI455X (gfx1250) — matrix operand order repaired and checked by reading, not compiled, not run on hardware
//
#include <hip/hip_runtime.h>
#include <hip/hip_bf16.h>
#include <stdint.h>

// ---------------------------------------------------------------------------
// Int4 fake-quant tiny LSTM for MI455X (gfx1250, wave32, WMMA f16)
//   B=16, S=512, D=256, L=2, VOCAB=32000, G=4D=1024
// ---------------------------------------------------------------------------

typedef __attribute__((ext_vector_type(16))) _Float16 v16h;
typedef __attribute__((ext_vector_type(8)))  _Float16 v8h;
typedef __attribute__((ext_vector_type(8)))  float    v8f;

#define B_  16
#define S_  512
#define D_  256
#define L_  2
#define V_  32000
#define G_  1024          // 4*D
#define KT_ 8             // K tiles of 32 along D=256

// ---------------- WMMA helpers -------------------------------------------

__device__ __forceinline__ v8f wmma_f16(v16h a, v16h b, v8f c) {
  return __builtin_amdgcn_wmma_f32_16x16x32_f16(
      /*neg_a=*/false, a, /*neg_b=*/false, b,
      /*c_mod=*/(short)0, c, /*reuse_a=*/false, /*reuse_b=*/false);
}

// A-fragment (16x32, f16) from row-major [16 x D_] GLOBAL source.
// ISA 7.12.2 16-bit A layout: lanes 0-15 -> M=lane, K chunks {kb..kb+7, kb+16..kb+23};
// lanes 16-31 -> M=lane-16, K chunks {kb+8..kb+15, kb+24..kb+31}.
__device__ __forceinline__ v16h load_a_frag_g(const _Float16* __restrict__ src,
                                              int kbase, int lane) {
  int row  = lane & 15;
  int koff = (lane >> 4) * 8;
  const _Float16* p = src + (size_t)row * D_ + kbase + koff;
  v8h lo = *(const v8h*)(p);
  v8h hi = *(const v8h*)(p + 16);
  return __builtin_shufflevector(lo, hi, 0,1,2,3,4,5,6,7,8,9,10,11,12,13,14,15);
}

// Same A-fragment layout, but sourced from an LDS-resident [B_][D_] buffer.
// Direct indexing of the __shared__ array keeps the pointer chain in AS(3)
// so these lower to ds_load_b128 (DScnt) instead of flat loads.
__device__ __forceinline__ v16h load_a_frag_sh(const _Float16 (&hb)[B_][D_],
                                               int kbase, int lane) {
  int row  = lane & 15;
  int koff = (lane >> 4) * 8;
  const _Float16* p = &hb[row][kbase + koff];
  v8h lo = *(const v8h*)(p);
  v8h hi = *(const v8h*)(p + 16);
  return __builtin_shufflevector(lo, hi, 0,1,2,3,4,5,6,7,8,9,10,11,12,13,14,15);
}

// B-fragment from pre-packed buffer: tile-major, 32 lanes x 16 halves contiguous.
__device__ __forceinline__ v16h load_b_frag(const _Float16* __restrict__ packed,
                                            long long tile, int lane) {
  const _Float16* p = packed + ((size_t)tile * 32 + (size_t)lane) * 16;
  v8h lo = *(const v8h*)(p);
  v8h hi = *(const v8h*)(p + 8);
  return __builtin_shufflevector(lo, hi, 0,1,2,3,4,5,6,7,8,9,10,11,12,13,14,15);
}

__device__ __forceinline__ float sigmoidf_(float x) { return 1.0f / (1.0f + __expf(-x)); }

__device__ __forceinline__ float quant4(float w, float scale, float inv) {
  float q = rintf(w * inv);
  q = fminf(fmaxf(q, -8.0f), 7.0f);
  return q * scale;
}

// ---------------- prologue kernels ---------------------------------------

__global__ void i4lstm_init_slots(unsigned* slots) {
  if (threadIdx.x < 8) slots[threadIdx.x] = 0u;
}

__global__ void i4lstm_absmax(const float* __restrict__ src, long long n, unsigned* slot) {
  __shared__ float red[256];
  float m = 0.0f;
  for (long long i = (long long)blockIdx.x * blockDim.x + threadIdx.x; i < n;
       i += (long long)gridDim.x * blockDim.x)
    m = fmaxf(m, fabsf(src[i]));
  red[threadIdx.x] = m;
  __syncthreads();
  for (int s = 128; s > 0; s >>= 1) {
    if ((int)threadIdx.x < s) red[threadIdx.x] = fmaxf(red[threadIdx.x], red[threadIdx.x + s]);
    __syncthreads();
  }
  if (threadIdx.x == 0) atomicMax(slot, __float_as_uint(red[0]));
}

// Fake-quantize src [J x K=256] (row-major) into WMMA-B packed f16 fragments.
// Lane n<16 carries column (jt*16+n), K=kt*32+{0..7,16..23}; lane n+16 carries K=+{8..15,24..31}.
__global__ void i4lstm_pack_b(const float* __restrict__ src, const unsigned* __restrict__ slot,
                              _Float16* __restrict__ dst, long long nthreads) {
  long long id = (long long)blockIdx.x * blockDim.x + threadIdx.x;
  if (id >= nthreads) return;
  int lane = (int)(id & 31);
  long long tile = id >> 5;
  long long jt = tile / KT_;
  int kt = (int)(tile % KT_);
  float am    = __uint_as_float(*slot);
  float scale = fmaxf(am / 7.0f, 1e-8f);
  float inv   = 1.0f / scale;
  int n  = lane & 15;
  int ks = (lane >> 4) * 8;
  const float* sp = src + (jt * 16 + n) * (long long)D_ + kt * 32 + ks;
  _Float16* dp = dst + id * 16;
  #pragma unroll
  for (int i = 0; i < 16; ++i) dp[i] = (_Float16)quant4(sp[i + (i >> 3) * 8], scale, inv);
}

// Embedding gather + fake-quant -> time-major f16 x_seq[s][b][d]
__global__ void i4lstm_gather(const int* __restrict__ x, const float* __restrict__ emb,
                              const unsigned* __restrict__ slot, _Float16* __restrict__ xseq) {
  long long id = (long long)blockIdx.x * blockDim.x + threadIdx.x;
  if (id >= (long long)S_ * B_ * D_) return;
  int d = (int)(id & (D_ - 1));
  long long r = id >> 8;
  int b = (int)(r & (B_ - 1));
  int s = (int)(r >> 4);
  int tok = x[b * S_ + s];
  float am    = __uint_as_float(*slot);
  float scale = fmaxf(am / 7.0f, 1e-8f);
  float inv   = 1.0f / scale;
  float w = emb[(long long)tok * D_ + d];
  xseq[id] = (_Float16)quant4(w, scale, inv);
}

// ---------------- persistent LSTM kernel (single WGP, 32 waves) ----------

__global__ __launch_bounds__(1024) void i4lstm_lstm(
    const _Float16* __restrict__ xseq,   // [S][B][D] f16
    const _Float16* __restrict__ wihP,   // packed, [L][512 tiles][32][16]
    const _Float16* __restrict__ whhP,   // packed, [L][512 tiles][32][16]
    const float* __restrict__ b_ih,      // [L][1024]
    const float* __restrict__ b_hh,      // [L][1024]
    _Float16* __restrict__ lstm_out)     // [S][B][D] f16
{
  __shared__ float    gates[B_][G_];       // 64 KB
  __shared__ _Float16 hbuf[L_][B_][D_];    // 16 KB
  __shared__ float    cbuf[L_][B_][D_];    // 32 KB

  const int tid  = threadIdx.x;
  const int lane = tid & 31;
  const int wave = tid >> 5;

  for (int i = tid; i < L_ * B_ * D_; i += 1024) {
    ((_Float16*)hbuf)[i] = (_Float16)0.0f;
    ((float*)cbuf)[i]    = 0.0f;
  }
  __syncthreads();

  for (int t = 0; t < S_; ++t) {
    if (t + 1 < S_)
      __builtin_prefetch((const void*)(xseq + (size_t)(t + 1) * (B_ * D_)), 0, 1);

    for (int l = 0; l < L_; ++l) {
      // Launder the packed-weight OFFSET (not the pointer!) so the compiler
      // cannot hoist the t-invariant B-fragment loads out of the timestep
      // loop (which spilled to scratch), while the loads keep their
      // kernel-arg provenance and lower to global_load_b128 (not flat).
      size_t woff = (size_t)l * (512 * 512);
      asm volatile("" : "+s"(woff));
      const _Float16* wp = wihP + woff;
      const _Float16* rp = whhP + woff;

      // Each wave owns 2 adjacent N-tiles; one shared A-fragment per k-step
      // feeds both tiles (halves the ds_load/A traffic vs. jj-outer order).
      const long long jt0 = (long long)wave * 2;
      const long long jt1 = jt0 + 1;
      v8f acc0 = {};
      v8f acc1 = {};

      if (l == 0) {
        const _Float16* ag = xseq + (size_t)t * (B_ * D_);
        #pragma unroll
        for (int kk = 0; kk < KT_; ++kk) {            // input GEMM: A = x_t (global)
          v16h a  = load_a_frag_g(ag, kk * 32, lane);
          acc0 = wmma_f16(a, load_b_frag(wp, jt0 * KT_ + kk, lane), acc0);
          acc1 = wmma_f16(a, load_b_frag(wp, jt1 * KT_ + kk, lane), acc1);
        }
        #pragma unroll
        for (int kk = 0; kk < KT_; ++kk) {            // recurrent GEMM: A = h0(t-1) (LDS)
          v16h a  = load_a_frag_sh(hbuf[0], kk * 32, lane);
          acc0 = wmma_f16(a, load_b_frag(rp, jt0 * KT_ + kk, lane), acc0);
          acc1 = wmma_f16(a, load_b_frag(rp, jt1 * KT_ + kk, lane), acc1);
        }
      } else {
        #pragma unroll
        for (int kk = 0; kk < KT_; ++kk) {            // input GEMM: A = h0(t) (LDS, fresh)
          v16h a  = load_a_frag_sh(hbuf[0], kk * 32, lane);
          acc0 = wmma_f16(a, load_b_frag(wp, jt0 * KT_ + kk, lane), acc0);
          acc1 = wmma_f16(a, load_b_frag(wp, jt1 * KT_ + kk, lane), acc1);
        }
        #pragma unroll
        for (int kk = 0; kk < KT_; ++kk) {            // recurrent GEMM: A = h1(t-1) (LDS)
          v16h a  = load_a_frag_sh(hbuf[1], kk * 32, lane);
          acc0 = wmma_f16(a, load_b_frag(rp, jt0 * KT_ + kk, lane), acc0);
          acc1 = wmma_f16(a, load_b_frag(rp, jt1 * KT_ + kk, lane), acc1);
        }
      }

      {
        const int n  = lane & 15;
        const int mh = (lane >> 4) * 8;
        #pragma unroll
        for (int r = 0; r < 8; ++r) {
          gates[mh + r][(int)jt0 * 16 + n] = acc0[r];
          gates[mh + r][(int)jt1 * 16 + n] = acc1[r];
        }
      }
      __syncthreads();

      // Elementwise gate combine + state update (4096 (b,d) pairs).
      for (int u = tid; u < B_ * D_; u += 1024) {
        const int b = u >> 8;
        const int d = u & (D_ - 1);
        const float bi = b_ih[l * G_ + d]            + b_hh[l * G_ + d];
        const float bf = b_ih[l * G_ + D_ + d]       + b_hh[l * G_ + D_ + d];
        const float bg = b_ih[l * G_ + 2 * D_ + d]   + b_hh[l * G_ + 2 * D_ + d];
        const float bo = b_ih[l * G_ + 3 * D_ + d]   + b_hh[l * G_ + 3 * D_ + d];
        const float ig = sigmoidf_(gates[b][d] + bi);
        const float fg = sigmoidf_(gates[b][D_ + d] + bf);
        const float gg = tanhf(gates[b][2 * D_ + d] + bg);
        const float og = sigmoidf_(gates[b][3 * D_ + d] + bo);
        const float c  = fg * cbuf[l][b][d] + ig * gg;
        const float h  = og * tanhf(c);
        cbuf[l][b][d] = c;
        hbuf[l][b][d] = (_Float16)h;
        if (l == L_ - 1)
          lstm_out[((size_t)t * B_ + b) * D_ + d] = (_Float16)h;
      }
      __syncthreads();
    }
  }
}

// ---------------- output projection GEMM ---------------------------------
// C[b*S+s, col] = lstm_out[s][b][:] . fcw_q[col][:] + fc_b[col]
// grid = (2000/8, 512), block = 256 (8 waves); each wave: one 16x16 tile.
// Logits (1.05 GB, write-once) use non-temporal stores so they do not evict
// the L2-resident packed weights.

__global__ __launch_bounds__(256) void i4lstm_fc(
    const _Float16* __restrict__ aIn,   // [S][B][D] f16
    const _Float16* __restrict__ fcwP,  // packed [2000*8 tiles][32][16]
    const float* __restrict__ fc_b,     // [V]
    float* __restrict__ out)            // [B][S][V] f32
{
  const int lane = threadIdx.x & 31;
  const int wave = threadIdx.x >> 5;
  const int jt   = blockIdx.x * 8 + wave;   // 0..1999
  const int s    = blockIdx.y;              // 0..511
  const _Float16* A = aIn + (size_t)s * (B_ * D_);

  v8f acc = {};
  #pragma unroll
  for (int kk = 0; kk < KT_; ++kk) {
    v16h a = load_a_frag_g(A, kk * 32, lane);
    v16h b = load_b_frag(fcwP, (long long)jt * KT_ + kk, lane);
    acc = wmma_f16(a, b, acc);
  }

  const int n   = lane & 15;
  const int col = jt * 16 + n;
  const int mh  = (lane >> 4) * 8;
  const float bias = fc_b[col];
  #pragma unroll
  for (int r = 0; r < 8; ++r) {
    const int b = mh + r;   // batch index
    __builtin_nontemporal_store(acc[r] + bias, &out[((size_t)b * S_ + s) * V_ + col]);
  }
}

// ---------------- host-side orchestration --------------------------------

extern "C" void kernel_launch(void* const* d_in, const int* in_sizes, int n_in,
                              void* d_out, int out_size, void* d_ws, size_t ws_size,
                              hipStream_t stream) {
  const int*   x     = (const int*)  d_in[0];   // [B,S]
  const float* emb_w = (const float*)d_in[1];   // [V,D]
  const float* w_ih  = (const float*)d_in[2];   // [L,4D,D]
  const float* b_ih  = (const float*)d_in[3];   // [L,4D]
  const float* w_hh  = (const float*)d_in[4];   // [L,4D,D]
  const float* b_hh  = (const float*)d_in[5];   // [L,4D]
  const float* fc_w  = (const float*)d_in[6];   // [V,D]
  const float* fc_b  = (const float*)d_in[7];   // [V]
  float* out = (float*)d_out;                   // [B,S,V]

  uint8_t* ws = (uint8_t*)d_ws;
  unsigned* slots = (unsigned*)ws;                         // [0]=emb [1]=fc [2,3]=wih [4,5]=whh
  _Float16* xseq     = (_Float16*)(ws + 256);              // S*B*D halves  (4 MB)
  _Float16* lstm_out = xseq + (size_t)S_ * B_ * D_;        // S*B*D halves  (4 MB)
  _Float16* wihP     = lstm_out + (size_t)S_ * B_ * D_;    // L*512*512 halves (1 MB)
  _Float16* whhP     = wihP + (size_t)L_ * 512 * 512;      // (1 MB)
  _Float16* fcwP     = whhP + (size_t)L_ * 512 * 512;      // 2000*8*512 halves (16 MB)

  // 1) zero absmax slots (deterministic per call)
  hipLaunchKernelGGL(i4lstm_init_slots, dim3(1), dim3(32), 0, stream, slots);

  // 2) per-tensor absmax reductions
  const long long nBig = (long long)V_ * D_;       // 8,192,000
  const long long nRec = (long long)G_ * D_;       // 262,144 per layer
  hipLaunchKernelGGL(i4lstm_absmax, dim3(1024), dim3(256), 0, stream, emb_w, nBig, slots + 0);
  hipLaunchKernelGGL(i4lstm_absmax, dim3(1024), dim3(256), 0, stream, fc_w,  nBig, slots + 1);
  for (int l = 0; l < L_; ++l) {
    hipLaunchKernelGGL(i4lstm_absmax, dim3(128), dim3(256), 0, stream,
                       w_ih + (size_t)l * nRec, nRec, slots + 2 + l);
    hipLaunchKernelGGL(i4lstm_absmax, dim3(128), dim3(256), 0, stream,
                       w_hh + (size_t)l * nRec, nRec, slots + 4 + l);
  }

  // 3) fake-quantize + pack weights into WMMA B-fragment layout
  const long long fcThreads  = (long long)(V_ / 16) * KT_ * 32;   // 512,000
  const long long recThreads = (long long)(G_ / 16) * KT_ * 32;   // 16,384
  hipLaunchKernelGGL(i4lstm_pack_b, dim3((unsigned)((fcThreads + 255) / 256)), dim3(256), 0, stream,
                     fc_w, slots + 1, fcwP, fcThreads);
  for (int l = 0; l < L_; ++l) {
    hipLaunchKernelGGL(i4lstm_pack_b, dim3((unsigned)((recThreads + 255) / 256)), dim3(256), 0, stream,
                       w_ih + (size_t)l * nRec, slots + 2 + l, wihP + (size_t)l * 512 * 512, recThreads);
    hipLaunchKernelGGL(i4lstm_pack_b, dim3((unsigned)((recThreads + 255) / 256)), dim3(256), 0, stream,
                       w_hh + (size_t)l * nRec, slots + 4 + l, whhP + (size_t)l * 512 * 512, recThreads);
  }

  // 4) embedding gather (+fake-quant) into time-major f16
  const long long gThreads = (long long)S_ * B_ * D_;             // 2,097,152
  hipLaunchKernelGGL(i4lstm_gather, dim3((unsigned)((gThreads + 255) / 256)), dim3(256), 0, stream,
                     x, emb_w, slots + 0, xseq);

  // 5) persistent recurrent kernel: 1 workgroup, 32 wave32 waves, state in LDS
  hipLaunchKernelGGL(i4lstm_lstm, dim3(1), dim3(1024), 0, stream,
                     xseq, wihP, whhP, b_ih, b_hh, lstm_out);

  // 6) output projection: [8192 x 256] x [256 x 32000] via WMMA f16
  hipLaunchKernelGGL(i4lstm_fc, dim3(V_ / 16 / 8, S_), dim3(256), 0, stream,
                     lstm_out, fcwP, fc_b, out);
}
